// MoMAttentionSelf_78391743086662
// MI455X (gfx1250) — matrix operand order repaired and checked by reading, not compiled, not run on hardware
//
#include <hip/hip_runtime.h>
#include <hip/hip_bf16.h>

// ---------------------------------------------------------------------------
// MoM gated delta-rule layer for MI455X (gfx1250, wave32).
// - Projections: LDS-tiled bf16 WMMA GEMM (64x64 tile / 4 waves), K-panels
//   staged via the Tensor Data Mover (tensor_load_to_lds + s_wait_tensorcnt).
// - Scan: register-resident VALU recurrence, unrouted tokens skipped
//   uniformly, expert combine via global f32 atomics.
// ---------------------------------------------------------------------------

typedef __attribute__((ext_vector_type(16))) __bf16 v16bf;
typedef __attribute__((ext_vector_type(8)))  float  v8f;
typedef __attribute__((ext_vector_type(4)))  unsigned int u32x4;
typedef __attribute__((ext_vector_type(4)))  unsigned int v4u;
typedef __attribute__((ext_vector_type(8)))  int v8i;
typedef __attribute__((ext_vector_type(4)))  int v4i;

#if defined(__has_builtin)
#  if __has_builtin(__builtin_amdgcn_tensor_load_to_lds)
#    define HAVE_TDM 1
#  endif
#endif
#ifndef HAVE_TDM
#  define HAVE_TDM 0
#endif

#define E_    4
#define TOPK_ 2
#define H_    8
#define DK_   64
#define DV_   64
#define D_    512
#define B_    4
#define L_    1024
#define M_    (B_ * L_)      // 4096 tokens
#define VDIM_ (H_ * DV_)     // 512

__device__ __forceinline__ unsigned short f2bf(float x) {
  unsigned int u = __float_as_uint(x);
  unsigned int r = u + 0x7FFFu + ((u >> 16) & 1u);
  return (unsigned short)(r >> 16);
}
__device__ __forceinline__ float sigmoidf_(float x) { return 1.0f / (1.0f + __expf(-x)); }
__device__ __forceinline__ float siluf_(float x)    { return x * sigmoidf_(x); }
__device__ __forceinline__ float softplusf_(float x){ return (x > 20.f) ? x : log1pf(__expf(x)); }

union BfPack { v16bf v; u32x4 q[2]; };

// ---------------- conversion kernels ----------------
__global__ void k_cvt_bf16(const float* __restrict__ x, unsigned short* __restrict__ y, int n) {
  int i = blockIdx.x * blockDim.x + threadIdx.x;
  if (i < n) y[i] = f2bf(x[i]);
}

// W is [K x N] row-major f32 -> Wt is [N x K] row-major bf16
__global__ void k_cvt_transpose(const float* __restrict__ W, unsigned short* __restrict__ Wt,
                                int K, int N) {
  int i = blockIdx.x * blockDim.x + threadIdx.x;   // i = n*K + k
  if (i >= K * N) return;
  int n = i / K, k = i - n * K;
  Wt[i] = f2bf(W[(size_t)k * N + n]);
}

__global__ void k_zero(float* __restrict__ p, int n) {
  int i = blockIdx.x * blockDim.x + threadIdx.x;
  if (i < n) p[i] = 0.0f;
}

// ---------------- TDM: async-load one [64 x 32] bf16 tile into LDS ----------------
// Builds a Tensor DMA Descriptor (D#) per CDNA5 ISA:
//   group0: count=1 | lds_addr | global_addr | type=2
//   group1: data_size=2B, tensor_dim0=K, tensor_dim1=rows, tile=32x64, stride0=K
__device__ __forceinline__ void tdm_load_tile(unsigned lds_off, const unsigned short* gptr,
                                              int K, int rows) {
#if HAVE_TDM
  unsigned long long ga = (unsigned long long)(size_t)gptr;
  v4u g0 = { 1u,                                   // count=1, is_restore=0
             lds_off,                              // lds_addr (bytes)
             (unsigned)(ga & 0xFFFFFFFFu),         // global_addr[31:0]
             (unsigned)((ga >> 32) & 0x1FFFFFFu) | 0x80000000u };  // addr[56:32] | type=2
  unsigned K_u = (unsigned)K, R_u = (unsigned)rows;
  v8i g1 = { (int)0x00010000u,                         // data_size = 1 (2 bytes)
             (int)((K_u & 0xFFFFu) << 16),             // tensor_dim0[15:0]
             (int)((K_u >> 16) | ((R_u & 0xFFFFu) << 16)),  // td0[31:16] | td1[15:0]
             (int)((R_u >> 16) | (32u << 16)),         // td1[31:16] | tile_dim0=32
             (int)64,                                  // tile_dim1=64, tile_dim2=0
             (int)K_u,                                 // tensor_dim0_stride[31:0]
             0, 0 };
  v4i gz = {0, 0, 0, 0};
#if __clang_major__ >= 23
  v8i gz8 = {0, 0, 0, 0, 0, 0, 0, 0};
  __builtin_amdgcn_tensor_load_to_lds(g0, g1, gz, gz, gz8, 0);
#else
  __builtin_amdgcn_tensor_load_to_lds(g0, g1, gz, gz, 0);
#endif
#else
  (void)lds_off; (void)gptr; (void)K; (void)rows;
#endif
}

__device__ __forceinline__ unsigned lds_offset_of(const void* p) {
  return (unsigned)(size_t)(__attribute__((address_space(3))) const void*)p;
}

// ---------------- bf16 WMMA GEMM: C[MxN] = A[MxK] * Bt[NxK]^T ----------------
// 128 threads (4 waves) per 64x64 tile; K-panels staged in LDS (TDM if
// available), each wave computes a 16x64 strip = 4 accumulators.
__global__ void __launch_bounds__(128)
k_gemm_bf16(const unsigned short* __restrict__ A,
            const unsigned short* __restrict__ Bt,
            float* __restrict__ C, int M, int N, int K) {
  __shared__ unsigned short tileA[64 * 32];
  __shared__ unsigned short tileB[64 * 32];
  const int nb   = blockIdx.x * 64;
  const int mb   = blockIdx.y * 64;
  const int tid  = threadIdx.x;      // 0..127
  const int wave = tid >> 5;         // 0..3 -> M sub-tile
  const int lane = tid & 31;
  const int hi   = lane >> 4;        // K half-select (per ISA 16-bit striping)
  const int lo   = lane & 15;        // row (A) / column (B) index

  v8f acc[4] = {};
  for (int kk = 0; kk < K; kk += 32) {
#if HAVE_TDM
    if (tid < 32) {   // wave 0 issues both DMAs, waits TENSORcnt, then barrier
      tdm_load_tile(lds_offset_of(tileA), A  + (size_t)mb * K + kk, K, M - mb);
      tdm_load_tile(lds_offset_of(tileB), Bt + (size_t)nb * K + kk, K, N - nb);
      __builtin_amdgcn_s_wait_tensorcnt(0);
    }
    __syncthreads();
#else
    {
      int row = tid >> 1, half = tid & 1;
      const u32x4* ga = reinterpret_cast<const u32x4*>(A  + (size_t)(mb + row) * K + kk + half * 16);
      const u32x4* gb = reinterpret_cast<const u32x4*>(Bt + (size_t)(nb + row) * K + kk + half * 16);
      u32x4* la = reinterpret_cast<u32x4*>(tileA + row * 32 + half * 16);
      u32x4* lb = reinterpret_cast<u32x4*>(tileB + row * 32 + half * 16);
      la[0] = ga[0]; la[1] = ga[1];
      lb[0] = gb[0]; lb[1] = gb[1];
      __syncthreads();
    }
#endif
    // A lane fragment: elems 0..7 -> K=hi*8+i ; elems 8..15 -> K=16+hi*8+i
    BfPack a;
    const unsigned short* arow = tileA + (wave * 16 + lo) * 32;
    a.q[0] = *reinterpret_cast<const u32x4*>(arow + hi * 8);
    a.q[1] = *reinterpret_cast<const u32x4*>(arow + 16 + hi * 8);
#pragma unroll
    for (int nt = 0; nt < 4; ++nt) {   // reuse A fragment across 4 N sub-tiles
      BfPack b;
      const unsigned short* brow = tileB + (nt * 16 + lo) * 32;
      b.q[0] = *reinterpret_cast<const u32x4*>(brow + hi * 8);
      b.q[1] = *reinterpret_cast<const u32x4*>(brow + 16 + hi * 8);
      acc[nt] = __builtin_amdgcn_wmma_f32_16x16x32_bf16(
          false, a.v, false, b.v, (short)0, acc[nt], false, false);
    }
    __syncthreads();
  }
  // C/D layout: VGPR r -> M = r + hi*8 ; N = lane&15
#pragma unroll
  for (int nt = 0; nt < 4; ++nt)
#pragma unroll
    for (int r = 0; r < 8; ++r)
      C[(size_t)(mb + wave * 16 + r + hi * 8) * N + nb + nt * 16 + lo] = acc[nt][r];
}

// ---------------- silu + l2norm over DK per (token, head), in place ----------------
__global__ void k_silu_l2norm(float* __restrict__ x) {
  __shared__ float red[64];
  const int bid = blockIdx.x;           // m*H + h
  const int m = bid >> 3, h = bid & 7;
  const int tid = threadIdx.x;          // 0..63
  float* row = x + (size_t)m * VDIM_ + h * DK_;
  float v = siluf_(row[tid]);
  red[tid] = v * v;
  __syncthreads();
  for (int s = 32; s > 0; s >>= 1) {
    if (tid < s) red[tid] += red[tid + s];
    __syncthreads();
  }
  row[tid] = v * rsqrtf(red[0] + 1e-6f);
}

__global__ void k_silu(float* __restrict__ x, int n) {
  int i = blockIdx.x * blockDim.x + threadIdx.x;
  if (i < n) x[i] = siluf_(x[i]);
}

// ---------------- router + beta + g per token ----------------
__global__ void k_router(const float* __restrict__ h, const float* __restrict__ Wgate,
                         const float* __restrict__ Wb, const float* __restrict__ Wa,
                         const float* __restrict__ A_log, const float* __restrict__ dt_bias,
                         float* __restrict__ wfull, float* __restrict__ beta,
                         float* __restrict__ gval) {
  int t = blockIdx.x * blockDim.x + threadIdx.x;
  if (t >= M_) return;
  const float* hr = h + (size_t)t * D_;
  float sg[E_] = {0.f, 0.f, 0.f, 0.f};
  float sb[H_] = {0.f}, sa[H_] = {0.f};
  for (int d = 0; d < D_; ++d) {
    float hv = hr[d];
#pragma unroll
    for (int e = 0; e < E_; ++e) sg[e] += hv * Wgate[d * E_ + e];
#pragma unroll
    for (int j = 0; j < H_; ++j) { sb[j] += hv * Wb[d * H_ + j]; sa[j] += hv * Wa[d * H_ + j]; }
  }
  float mx = sg[0];
#pragma unroll
  for (int e = 1; e < E_; ++e) mx = fmaxf(mx, sg[e]);
  float sum = 0.f, p[E_];
#pragma unroll
  for (int e = 0; e < E_; ++e) { p[e] = __expf(sg[e] - mx); sum += p[e]; }
#pragma unroll
  for (int e = 0; e < E_; ++e) p[e] /= sum;
  int i0 = 0;
#pragma unroll
  for (int e = 1; e < E_; ++e) if (p[e] > p[i0]) i0 = e;
  int i1 = (i0 == 0) ? 1 : 0;
#pragma unroll
  for (int e = 0; e < E_; ++e) if (e != i0 && p[e] > p[i1]) i1 = e;
  float rn = p[i0] + p[i1];
#pragma unroll
  for (int e = 0; e < E_; ++e) wfull[t * E_ + e] = 0.f;
  wfull[t * E_ + i0] = p[i0] / rn;
  wfull[t * E_ + i1] = p[i1] / rn;
#pragma unroll
  for (int j = 0; j < H_; ++j) {
    beta[t * H_ + j] = sigmoidf_(sb[j]);
    gval[t * H_ + j] = -__expf(A_log[j]) * softplusf_(sa[j] + dt_bias[j]);
  }
}

// ---------------- recurrent gated delta-rule scan ----------------
// One block per (e,b,h); 64 threads; thread tid owns S[:, tid] (64 f32 regs).
// Unrouted tokens leave the state untouched and contribute nothing to the
// combine, so the whole step is skipped (routing weight is block-uniform).
__global__ void __launch_bounds__(64)
k_scan(const float* __restrict__ qp, const float* __restrict__ kp,
       const float* __restrict__ vp, const float* __restrict__ beta,
       const float* __restrict__ gval, const float* __restrict__ wfull,
       float* __restrict__ core) {
  const int bx = blockIdx.x;                 // 0..127
  const int e  = bx & 3;
  const int b  = (bx >> 2) & 3;
  const int hh = bx >> 4;                    // 0..7
  const int tid = threadIdx.x;               // 0..63 -> DV column
  const float scale = 0.125f;                // DK^-0.5

  __shared__ float sk[DK_], sq[DK_];
  float s[DK_];
#pragma unroll
  for (int i = 0; i < DK_; ++i) s[i] = 0.f;

  for (int t = 0; t < L_; ++t) {
    const int m = b * L_ + t;
    const float w = wfull[m * E_ + e];       // uniform across the block
    if (!(w > 0.f)) continue;                // unrouted: state unchanged

    const size_t rowoff = (size_t)m * VDIM_ + hh * DV_;
    sk[tid] = kp[rowoff + tid];
    sq[tid] = qp[rowoff + tid];
    __syncthreads();

    const float bt  = beta[m * H_ + hh];
    const float dec = __expf(gval[m * H_ + hh]);

    float kv = 0.f;
#pragma unroll
    for (int i = 0; i < DK_; ++i) { s[i] *= dec; kv = fmaf(s[i], sk[i], kv); }
    const float vt = vp[rowoff + tid];
    const float delta = (vt - kv) * bt;
    float o = 0.f;
#pragma unroll
    for (int i = 0; i < DK_; ++i) { s[i] = fmaf(sk[i], delta, s[i]); o = fmaf(sq[i], s[i], o); }

    atomicAdd(&core[rowoff + tid], w * o * scale);
    __syncthreads();
  }
}

// ---------------- fused RMSNorm + silu gate -> bf16 ----------------
__global__ void k_norm_gate(const float* __restrict__ core, const float* __restrict__ gp,
                            const float* __restrict__ o_norm_w,
                            unsigned short* __restrict__ out_bf16) {
  __shared__ float red[64];
  const int bid = blockIdx.x;           // m*H + h
  const int m = bid >> 3, h = bid & 7;
  const int tid = threadIdx.x;          // 0..63
  const size_t off = (size_t)m * VDIM_ + h * DV_;
  float c = core[off + tid];
  red[tid] = c * c;
  __syncthreads();
  for (int s = 32; s > 0; s >>= 1) {
    if (tid < s) red[tid] += red[tid + s];
    __syncthreads();
  }
  float r = rsqrtf(red[0] * (1.0f / DV_) + 1e-5f);
  float gate = siluf_(gp[off + tid]);
  out_bf16[off + tid] = f2bf(c * r * o_norm_w[tid] * gate);
}

// ---------------------------------------------------------------------------
extern "C" void kernel_launch(void* const* d_in, const int* in_sizes, int n_in,
                              void* d_out, int out_size, void* d_ws, size_t ws_size,
                              hipStream_t stream) {
  const float* h        = (const float*)d_in[0];
  const float* Wq       = (const float*)d_in[1];
  const float* Wgate    = (const float*)d_in[2];
  const float* Wk       = (const float*)d_in[3];
  const float* Wv       = (const float*)d_in[4];
  const float* Wb       = (const float*)d_in[5];
  const float* Wa       = (const float*)d_in[6];
  const float* A_log    = (const float*)d_in[7];
  const float* dt_bias  = (const float*)d_in[8];
  const float* Wg       = (const float*)d_in[9];
  const float* o_norm_w = (const float*)d_in[10];
  const float* Wo       = (const float*)d_in[11];
  float* out            = (float*)d_out;

  // ---- workspace carve-up ----
  char* ws = (char*)d_ws;
  size_t off = 0;
  auto alloc = [&](size_t bytes) { void* p = ws + off; off += (bytes + 255) & ~(size_t)255; return p; };

  unsigned short* hb    = (unsigned short*)alloc((size_t)M_ * D_ * 2);
  unsigned short* Wq_t  = (unsigned short*)alloc((size_t)D_ * VDIM_ * 2);
  unsigned short* Wk_t  = (unsigned short*)alloc((size_t)D_ * VDIM_ * 2);
  unsigned short* Wv_t  = (unsigned short*)alloc((size_t)D_ * VDIM_ * 2);
  unsigned short* Wg_t  = (unsigned short*)alloc((size_t)D_ * VDIM_ * 2);
  unsigned short* Wo_t  = (unsigned short*)alloc((size_t)VDIM_ * D_ * 2);
  float* qp    = (float*)alloc((size_t)M_ * VDIM_ * 4);
  float* kp    = (float*)alloc((size_t)M_ * VDIM_ * 4);
  float* vp    = (float*)alloc((size_t)M_ * VDIM_ * 4);
  float* gp    = (float*)alloc((size_t)M_ * VDIM_ * 4);
  float* beta  = (float*)alloc((size_t)M_ * H_ * 4);
  float* gval  = (float*)alloc((size_t)M_ * H_ * 4);
  float* wfull = (float*)alloc((size_t)M_ * E_ * 4);
  float* core  = (float*)alloc((size_t)M_ * VDIM_ * 4);
  unsigned short* coreb = (unsigned short*)alloc((size_t)M_ * VDIM_ * 2);
  (void)ws_size; (void)in_sizes; (void)n_in; (void)out_size;

  // ---- stage 1: conversions ----
  {
    int n = M_ * D_;
    k_cvt_bf16<<<(n + 255) / 256, 256, 0, stream>>>(h, hb, n);
  }
  {
    int n = D_ * VDIM_;
    dim3 g((n + 255) / 256);
    k_cvt_transpose<<<g, 256, 0, stream>>>(Wq, Wq_t, D_, VDIM_);
    k_cvt_transpose<<<g, 256, 0, stream>>>(Wk, Wk_t, D_, VDIM_);
    k_cvt_transpose<<<g, 256, 0, stream>>>(Wv, Wv_t, D_, VDIM_);
    k_cvt_transpose<<<g, 256, 0, stream>>>(Wg, Wg_t, D_, VDIM_);
    k_cvt_transpose<<<g, 256, 0, stream>>>(Wo, Wo_t, VDIM_, D_);
  }

  // ---- stage 2: WMMA projections (64x64 tiles, 4 waves/block) ----
  {
    dim3 grid(VDIM_ / 64, M_ / 64);   // (8, 64)
    k_gemm_bf16<<<grid, 128, 0, stream>>>(hb, Wq_t, qp, M_, VDIM_, D_);
    k_gemm_bf16<<<grid, 128, 0, stream>>>(hb, Wk_t, kp, M_, VDIM_, D_);
    k_gemm_bf16<<<grid, 128, 0, stream>>>(hb, Wv_t, vp, M_, VDIM_, D_);
    k_gemm_bf16<<<grid, 128, 0, stream>>>(hb, Wg_t, gp, M_, VDIM_, D_);
  }

  // ---- stage 3: epilogues + router ----
  k_silu_l2norm<<<M_ * H_, 64, 0, stream>>>(qp);
  k_silu_l2norm<<<M_ * H_, 64, 0, stream>>>(kp);
  {
    int n = M_ * VDIM_;
    k_silu<<<(n + 255) / 256, 256, 0, stream>>>(vp, n);
  }
  k_router<<<(M_ + 255) / 256, 256, 0, stream>>>(h, Wgate, Wb, Wa, A_log, dt_bias,
                                                 wfull, beta, gval);

  // ---- stage 4: scan with weighted combine (core must start at zero) ----
  {
    int n = M_ * VDIM_;
    k_zero<<<(n + 255) / 256, 256, 0, stream>>>(core, n);
  }
  k_scan<<<E_ * B_ * H_, 64, 0, stream>>>(qp, kp, vp, beta, gval, wfull, core);

  // ---- stage 5: RMSNorm + gate -> bf16, then output GEMM ----
  k_norm_gate<<<M_ * H_, 64, 0, stream>>>(core, gp, o_norm_w, coreb);
  {
    dim3 grid(D_ / 64, M_ / 64);
    k_gemm_bf16<<<grid, 128, 0, stream>>>(coreb, Wo_t, out, M_, D_, VDIM_);
  }
}
